// KPlanes_21320217658082
// MI455X (gfx1250) — matrix operand order repaired and checked by reading, not compiled, not run on hardware
//
#include <hip/hip_runtime.h>
#include <hip/hip_bf16.h>

typedef _Float16 v16h __attribute__((ext_vector_type(16)));
typedef _Float16 h8   __attribute__((ext_vector_type(8)));
typedef float    v8f  __attribute__((ext_vector_type(8)));

#define BP 128           // points per block
#define NTHREADS 256     // 8 wave32

// float-element offsets of each transposed plane inside the workspace
__constant__ unsigned int c_plane_off[12] = {
    0u,        524288u,   1048576u,
    1572864u,  3670016u,  5767168u,
    7864320u,  16252928u, 24641536u,
    33030144u, 66584576u, 100139008u
};

struct GridPtrs { const float* g[12]; };

// -------- pre-pass: [32,R,R] -> [R,R,32] so channel gathers are one cacheline --------
__global__ void kp_transpose32(const float* __restrict__ in, float* __restrict__ out, int R) {
  __shared__ float tile[32 * 33];
  const int tid = threadIdx.x;
  const int j = tid & 31;          // fast index
  const int g = tid >> 5;          // wave id 0..7
  const long long RR = (long long)R * R;
  const long long pix0 = (long long)blockIdx.x * 32;
#pragma unroll
  for (int i = 0; i < 4; ++i) {
    int c = g + i * 8;                                   // coalesced global read
    tile[c * 33 + j] = in[(long long)c * RR + pix0 + j];
  }
  __syncthreads();
#pragma unroll
  for (int i = 0; i < 4; ++i) {
    int jj = g + i * 8;                                  // coalesced global write
    out[(pix0 + jj) * 32 + j] = tile[j * 33 + jj];
  }
}

// -------- WMMA fragment loaders (ISA 16-bit A / B layouts, wave32) --------
__device__ __forceinline__ v16h load_a_frag(const _Float16* base, int stride,
                                            int m0, int k0, int nn, int hi) {
  // A halves 0..7  = row[k0 + 8*hi .. +8], halves 8..15 = row[k0 + 16 + 8*hi .. +8]
  const _Float16* p = base + (m0 + nn) * stride + k0 + 8 * hi;
  h8 c0 = *(const h8*)p;
  h8 c1 = *(const h8*)(p + 16);
  return __builtin_shufflevector(c0, c1, 0,1,2,3,4,5,6,7,8,9,10,11,12,13,14,15);
}
__device__ __forceinline__ v16h load_b_frag(const _Float16* wT, int stride,
                                            int n0, int k0, int nn, int hi) {
  // B halves 0..7 = W[k0 + 8*hi .. +8][n0+nn], halves 8..15 = W[k0 + 16 + 8*hi .. +8][n0+nn]; wT is N-major
  const _Float16* p = wT + (n0 + nn) * stride + k0 + 8 * hi;
  h8 c0 = *(const h8*)p;
  h8 c1 = *(const h8*)(p + 16);
  return __builtin_shufflevector(c0, c1, 0,1,2,3,4,5,6,7,8,9,10,11,12,13,14,15);
}

// -------- fused gather + MLP --------
__global__ void __launch_bounds__(NTHREADS)
kplanes_fused(const float* __restrict__ x, const float* __restrict__ tg, GridPtrs gp,
              int useT, const float* __restrict__ w0, const float* __restrict__ w1,
              const float* __restrict__ w2, float* __restrict__ out, int N)
{
  __shared__ _Float16 feat[BP * 128];   // features; cols 0..63 reused as h1 per-wave
  __shared__ _Float16 w0T[64 * 128];    // w0T[n][k] = w0[k][n], f16
  __shared__ _Float16 w1T[64 * 64];
  __shared__ float    xl[BP * 3];

  const int tid  = threadIdx.x;
  const int wave = tid >> 5;
  const int lane = tid & 31;
  const int nn = lane & 15;
  const int hi = lane >> 4;
  const long long p0 = (long long)blockIdx.x * BP;

  // ---- stage coords + weights ----
  for (int i = tid; i < BP * 3; i += NTHREADS) {
    long long gi = p0 * 3 + i;
    xl[i] = (gi < (long long)N * 3) ? x[gi] : 0.0f;
  }
  for (int i = tid; i < 128 * 64; i += NTHREADS) {
    int k = i >> 6, n = i & 63;
    w0T[n * 128 + k] = (_Float16)w0[i];
  }
  for (int i = tid; i < 64 * 64; i += NTHREADS) {
    int k = i >> 6, n = i & 63;
    w1T[n * 64 + k] = (_Float16)w1[i];
  }
  __syncthreads();

  // ---- gather: one wave-task = (point, scale); lane = channel ----
  for (int task = wave; task < BP * 4; task += 8) {
    const int p = task >> 2;
    const int s = task & 3;
    const int R = 128 << s;
    const size_t RR = (size_t)R * R;
    const float sc = 0.5f * (float)(R - 1);
    const float xv0 = xl[p * 3 + 0], xv1 = xl[p * 3 + 1], xv2 = xl[p * 3 + 2];
    float prod = 1.0f;
#pragma unroll
    for (int ci = 0; ci < 3; ++ci) {
      // PLANES = (0,1),(0,2),(1,2); cx -> W, cy -> H
      const float cx = (ci == 2) ? xv1 : xv0;
      const float cy = (ci == 0) ? xv1 : xv2;
      float fx = fminf(fmaxf((cx + 1.0f) * sc, 0.0f), (float)(R - 1));
      float fy = fminf(fmaxf((cy + 1.0f) * sc, 0.0f), (float)(R - 1));
      float x0f = floorf(fx), y0f = floorf(fy);
      float wx = fx - x0f,   wy = fy - y0f;
      int ix0 = (int)x0f, iy0 = (int)y0f;
      int ix1 = min(ix0 + 1, R - 1), iy1 = min(iy0 + 1, R - 1);
      size_t row0 = (size_t)iy0 * R, row1 = (size_t)iy1 * R;
      float g00, g01, g10, g11;
      if (useT) {  // [H,W,C]: one 128B line per corner across the wave
        const float* T = tg + (size_t)c_plane_off[s * 3 + ci];
        g00 = T[(row0 + ix0) * 32 + lane];
        g01 = T[(row0 + ix1) * 32 + lane];
        g10 = T[(row1 + ix0) * 32 + lane];
        g11 = T[(row1 + ix1) * 32 + lane];
      } else {     // fallback: original [C,H,W]
        const float* G = gp.g[s * 3 + ci] + (size_t)lane * RR;
        g00 = G[row0 + ix0];
        g01 = G[row0 + ix1];
        g10 = G[row1 + ix0];
        g11 = G[row1 + ix1];
      }
      float top = g00 + (g01 - g00) * wx;
      float bot = g10 + (g11 - g10) * wx;
      prod *= (top + (bot - top) * wy);
    }
    feat[p * 128 + s * 32 + lane] = (_Float16)prod;
  }
  __syncthreads();

  // ---- MLP: wave owns 16 points (rows m0..m0+15) ----
  const int m0 = wave * 16;

  // layer 0: [16x128] @ [128x64] -> relu -> h1 (into this wave's own feat rows, cols 0..63)
  v16h A0[4];
#pragma unroll
  for (int kt = 0; kt < 4; ++kt)
    A0[kt] = load_a_frag(feat, 128, m0, kt * 32, nn, hi);

#pragma unroll
  for (int nt = 0; nt < 4; ++nt) {
    v8f acc = {0.f, 0.f, 0.f, 0.f, 0.f, 0.f, 0.f, 0.f};
#pragma unroll
    for (int kt = 0; kt < 4; ++kt) {
      v16h B = load_b_frag(w0T, 128, nt * 16, kt * 32, nn, hi);
      acc = __builtin_amdgcn_wmma_f32_16x16x32_f16(false, A0[kt], false, B,
                                                   (short)0, acc, false, false);
    }
#pragma unroll
    for (int r = 0; r < 8; ++r) {  // D: VGPR r -> M = r + 8*hi, N = nn
      float v = fmaxf(acc[r], 0.0f);
      feat[(m0 + r + 8 * hi) * 128 + nt * 16 + nn] = (_Float16)v;
    }
  }

  // layer 1: [16x64] @ [64x64] -> relu, fused with layer 2 (dot with w2)
  v16h A1[2];
  A1[0] = load_a_frag(feat, 128, m0, 0, nn, hi);
  A1[1] = load_a_frag(feat, 128, m0, 32, nn, hi);

  float pm[8] = {0.f, 0.f, 0.f, 0.f, 0.f, 0.f, 0.f, 0.f};
#pragma unroll
  for (int nt = 0; nt < 4; ++nt) {
    v8f acc = {0.f, 0.f, 0.f, 0.f, 0.f, 0.f, 0.f, 0.f};
    v16h B0 = load_b_frag(w1T, 64, nt * 16, 0, nn, hi);
    acc = __builtin_amdgcn_wmma_f32_16x16x32_f16(false, A1[0], false, B0,
                                                 (short)0, acc, false, false);
    v16h B1 = load_b_frag(w1T, 64, nt * 16, 32, nn, hi);
    acc = __builtin_amdgcn_wmma_f32_16x16x32_f16(false, A1[1], false, B1,
                                                 (short)0, acc, false, false);
    const float w2v = w2[nt * 16 + nn];
#pragma unroll
    for (int r = 0; r < 8; ++r) {
      float v = fmaxf(acc[r], 0.0f) * w2v;   // relu then weight (f32)
      v += __shfl_xor(v, 1);                 // reduce over the 16 N-lanes
      v += __shfl_xor(v, 2);
      v += __shfl_xor(v, 4);
      v += __shfl_xor(v, 8);
      pm[r] += v;
    }
  }
  if (nn == 0) {
#pragma unroll
    for (int r = 0; r < 8; ++r) {
      long long gpnt = p0 + m0 + r + 8 * hi;
      if (gpnt < N) out[gpnt] = pm[r];
    }
  }
}

extern "C" void kernel_launch(void* const* d_in, const int* in_sizes, int n_in,
                              void* d_out, int out_size, void* d_ws, size_t ws_size,
                              hipStream_t stream) {
  const float* x  = (const float*)d_in[0];
  const float* w0 = (const float*)d_in[13];
  const float* w1 = (const float*)d_in[14];
  const float* w2 = (const float*)d_in[15];
  float* out = (float*)d_out;
  float* ws  = (float*)d_ws;
  const int N = in_sizes[0] / 3;

  static const unsigned int h_off[12] = {
      0u,        524288u,   1048576u,
      1572864u,  3670016u,  5767168u,
      7864320u,  16252928u, 24641536u,
      33030144u, 66584576u, 100139008u};
  const size_t need = 133693440ull * sizeof(float);   // all transposed planes
  const int useT = (d_ws != nullptr && ws_size >= need) ? 1 : 0;

  GridPtrs gp;
  for (int i = 0; i < 12; ++i) gp.g[i] = (const float*)d_in[1 + i];

  if (useT) {
    for (int idx = 0; idx < 12; ++idx) {
      int s = idx / 3;
      int R = 128 << s;
      int blocks = (R * R) / 32;
      kp_transpose32<<<blocks, NTHREADS, 0, stream>>>(gp.g[idx], ws + (size_t)h_off[idx], R);
    }
  }
  int nblk = (N + BP - 1) / BP;
  kplanes_fused<<<nblk, NTHREADS, 0, stream>>>(x, ws, gp, useT, w0, w1, w2, out, N);
}
